// VectorQuantizer_58755152609798
// MI455X (gfx1250) — matrix operand order repaired and checked by reading, not compiled, not run on hardware
//
#include <hip/hip_runtime.h>

typedef __attribute__((ext_vector_type(16))) __bf16 v16bf;
typedef __attribute__((ext_vector_type(8)))  __bf16 v8bf;
typedef __attribute__((ext_vector_type(8)))  float  v8f;

#define BQ  32
#define DQ  256
#define HWQ 1024
#define NQ  (BQ * HWQ)   // 32768 latent vectors
#define KQ  1024         // codebook entries

#define TPAD   264                   // padded LDS row stride (ushorts): 256 + 8
#define TILEUS (16 * TPAD)           // one padded tile (hi or lo), ushorts = 4224
#define BUFUS  (2 * TILEUS)          // hi+lo for one buffer = 8448 ushorts
#define SRED_US (2 * BUFUS)          // reduction area starts after 2 buffers (16896 us)
#define SMEM_BYTES (SRED_US * 2 + 8 * 32 * 8 * 4 * 2)   // 33792 + 16384 = 50176

// ---------- helpers: f32 -> bf16 (RNE) split ----------
__device__ __forceinline__ unsigned short f2bf_rn(float f) {
  unsigned u = __float_as_uint(f);
  u += 0x7FFFu + ((u >> 16) & 1u);
  return (unsigned short)(u >> 16);
}
__device__ __forceinline__ float bf2f(unsigned short h) {
  return __uint_as_float(((unsigned)h) << 16);
}

// ---------- stage 0: codebook -> bf16 hi/lo + ||e||^2 ----------
__global__ void vq_prep_weight(const float* __restrict__ weight,
                               unsigned short* __restrict__ whi,
                               unsigned short* __restrict__ wlo,
                               float* __restrict__ ek2) {
  __shared__ float red[256];
  int k = blockIdx.x;
  int d = threadIdx.x;
  float w = weight[k * DQ + d];
  unsigned short h = f2bf_rn(w);
  whi[k * DQ + d] = h;
  wlo[k * DQ + d] = f2bf_rn(w - bf2f(h));
  red[d] = w * w;
  __syncthreads();
  for (int s = 128; s > 0; s >>= 1) {
    if (d < s) red[d] += red[d + s];
    __syncthreads();
  }
  if (d == 0) ek2[k] = red[0];
}

// ---------- stage 0b: latent (B,D,HW) -> x (N,D) bf16 hi/lo, LDS transpose ----------
__global__ void vq_prep_latent(const float* __restrict__ latent,
                               unsigned short* __restrict__ xhi,
                               unsigned short* __restrict__ xlo) {
  __shared__ float tile[32][33];
  int b   = blockIdx.z;
  int d0  = blockIdx.y * 32;
  int hw0 = blockIdx.x * 32;
  int tx = threadIdx.x, ty = threadIdx.y;
  const float* src = latent + (size_t)b * DQ * HWQ;
#pragma unroll
  for (int it = 0; it < 4; ++it) {
    int dd = d0 + ty + it * 8;
    tile[ty + it * 8][tx] = src[(size_t)dd * HWQ + hw0 + tx];  // coalesced over hw
  }
  __syncthreads();
#pragma unroll
  for (int it = 0; it < 4; ++it) {
    int hwi = ty + it * 8;
    size_t row = (size_t)(b * HWQ + hw0 + hwi) * DQ + d0 + tx; // coalesced over d
    float v = tile[tx][hwi];
    unsigned short h = f2bf_rn(v);
    xhi[row] = h;
    xlo[row] = f2bf_rn(v - bf2f(h));
  }
}

// ---------- async stage of one 16-code tile (hi+lo) into an LDS buffer ----------
// Tile region in whi/wlo is contiguous: 16 codes * 256 d = 4096 ushorts = 8 KB each.
// 256 threads x 2 chunks x (hi,lo) = 4 async B128 loads per thread.
__device__ __forceinline__ void issue_tile_async(const unsigned short* __restrict__ whi,
                                                 const unsigned short* __restrict__ wlo,
                                                 int tile, int buf, int t) {
  const unsigned short* gh = whi + (size_t)tile * 4096;
  const unsigned short* gl = wlo + (size_t)tile * 4096;
  unsigned bufBase = (unsigned)(buf * BUFUS) * 2u;          // byte offset in LDS
#pragma unroll
  for (int j = 0; j < 2; ++j) {
    int c = t + j * 256;                                    // 16-byte chunk id, 0..511
    int row = c >> 5;                                       // code row within tile
    int col = c & 31;                                       // 8-ushort column chunk
    unsigned loff_h = bufBase + (unsigned)(row * TPAD + col * 8) * 2u;
    unsigned loff_l = loff_h + (unsigned)TILEUS * 2u;
    unsigned long long gah = (unsigned long long)(const void*)(gh + c * 8);
    unsigned long long gal = (unsigned long long)(const void*)(gl + c * 8);
    asm volatile("global_load_async_to_lds_b128 %0, %1, off"
                 :: "v"(loff_h), "v"(gah) : "memory");
    asm volatile("global_load_async_to_lds_b128 %0, %1, off"
                 :: "v"(loff_l), "v"(gal) : "memory");
  }
}

// B fragment load from LDS: bf16 [p..p+7] and [p+16..p+23] -> two ds_load_b128
__device__ __forceinline__ v16bf ldB(const unsigned short* p) {
  v8bf a = *(const v8bf*)(p);
  v8bf b = *(const v8bf*)(p + 16);
  return __builtin_shufflevector(a, b, 0,1,2,3,4,5,6,7,8,9,10,11,12,13,14,15);
}

// ---------- stage 1: compensated-bf16 WMMA distances + argmin ----------
// 8 waves / block, 16 rows / wave; B tiles staged in LDS once per block (8x L2 saving),
// double-buffered with ASYNCcnt; B fragments register-pipelined one k-step ahead;
// two independent accumulator chains (hi*hi) and (lo*hi + hi*lo).
__global__ void __launch_bounds__(256) vq_argmin(
    const unsigned short* __restrict__ xhi, const unsigned short* __restrict__ xlo,
    const unsigned short* __restrict__ whi, const unsigned short* __restrict__ wlo,
    const float* __restrict__ ek2, int* __restrict__ idxOut) {
  extern __shared__ unsigned short smem[];                  // dynamic LDS, base offset 0
  float* sval = (float*)(smem + SRED_US);                   // [8][32][8]
  int*   sidx = (int*)(smem + SRED_US) + 8 * 32 * 8;

  int t     = threadIdx.x;
  int lane  = t & 31;
  int wave  = t >> 5;
  int lrow  = lane & 15;   // A-matrix row M / B-matrix column N held by this lane
  int lhalf = lane >> 4;   // K-half select per ISA 16-bit layouts
  int n0 = blockIdx.x * 128 + wave * 16;

  // A fragments: ISA 16-bit A 16x32: lane(<16): K = d0..d0+7 and d0+16..d0+23
  //              lane(>=16): same + 8.   Held in VGPRs for the whole kernel.
  const unsigned short* arh = xhi + (size_t)(n0 + lrow) * DQ;
  const unsigned short* arl = xlo + (size_t)(n0 + lrow) * DQ;
  v16bf Ahi[8], Alo[8];
#pragma unroll
  for (int s = 0; s < 8; ++s) {
    int d0 = s * 32 + lhalf * 8;
    v8bf h0 = *(const v8bf*)(arh + d0);
    v8bf h1 = *(const v8bf*)(arh + d0 + 16);
    Ahi[s] = __builtin_shufflevector(h0, h1, 0,1,2,3,4,5,6,7,8,9,10,11,12,13,14,15);
    v8bf l0 = *(const v8bf*)(arl + d0);
    v8bf l1 = *(const v8bf*)(arl + d0 + 16);
    Alo[s] = __builtin_shufflevector(l0, l1, 0,1,2,3,4,5,6,7,8,9,10,11,12,13,14,15);
  }

  float best[8];
  int   bidx[8];
#pragma unroll
  for (int r = 0; r < 8; ++r) { best[r] = 3.4e38f; bidx[r] = 0; }

  // Prologue: prefetch tiles 0 and 1 into buffers 0 and 1.
  issue_tile_async(whi, wlo, 0, 0, t);
  issue_tile_async(whi, wlo, 1, 1, t);

  for (int tile = 0; tile < 64; ++tile) {
    int buf = tile & 1;
    // Wait for this tile's async loads (in-order: <=4 leaves only the next tile's).
    if (tile < 63) asm volatile("s_wait_asynccnt 0x4" ::: "memory");
    else           asm volatile("s_wait_asynccnt 0x0" ::: "memory");
    __syncthreads();   // all threads' portions of buf are in LDS

    // Per-lane B pointers: lane holds column code=tile*16+lrow, K = lhalf*8 + {0..7, 16..23}.
    const unsigned short* bh = smem + buf * BUFUS + lrow * TPAD + lhalf * 8;
    const unsigned short* bl = bh + TILEUS;

    v8f cm = {};   // hi*hi chain
    v8f cc = {};   // compensation chain: lo*hi + hi*lo
    // Register double-buffer: fragments for k-step s+1 load while s computes.
    v16bf Bh = ldB(bh);
    v16bf Bl = ldB(bl);
#pragma unroll
    for (int s = 0; s < 8; ++s) {
      v16bf Bhc = Bh, Blc = Bl;
      if (s < 7) {
        Bh = ldB(bh + (s + 1) * 32);
        Bl = ldB(bl + (s + 1) * 32);
      }
      cm = __builtin_amdgcn_wmma_f32_16x16x32_bf16(false, Ahi[s], false, Bhc, (short)0, cm, false, false);
      cc = __builtin_amdgcn_wmma_f32_16x16x32_bf16(false, Alo[s], false, Bhc, (short)0, cc, false, false);
      cc = __builtin_amdgcn_wmma_f32_16x16x32_bf16(false, Ahi[s], false, Blc, (short)0, cc, false, false);
    }

    int code = tile * 16 + lrow;
    float e2 = ek2[code];
#pragma unroll
    for (int r = 0; r < 8; ++r) {
      float dst = e2 - 2.0f * (cm[r] + cc[r]);  // ||x||^2 dropped (row constant)
      if (dst < best[r]) { best[r] = dst; bidx[r] = code; }  // strict <: first-min wins
    }

    __syncthreads();   // everyone done reading buf before it is overwritten
    if (tile + 2 < 64) issue_tile_async(whi, wlo, tile + 2, buf, t);
  }

  // cross-lane argmin: row M lives at vgpr r=M&7, lanes [0,16) for M<8 else [16,32)
#pragma unroll
  for (int r = 0; r < 8; ++r) {
    sval[(wave * 32 + lane) * 8 + r] = best[r];
    sidx[(wave * 32 + lane) * 8 + r] = bidx[r];
  }
  __syncthreads();
  if (lane < 16) {
    int M = lane, r = M & 7, lg = (M < 8) ? 0 : 16;
    float bv = 3.4e38f; int bi = 0x7fffffff;
    for (int tt = 0; tt < 16; ++tt) {
      float v = sval[(wave * 32 + lg + tt) * 8 + r];
      int  ii = sidx[(wave * 32 + lg + tt) * 8 + r];
      if (v < bv || (v == bv && ii < bi)) { bv = v; bi = ii; }
    }
    idxOut[n0 + M] = bi;
  }
}

// ---------- stage 2: gather exact f32 codes, transpose out, partial loss ----------
__global__ void vq_gather(const float* __restrict__ latent, const float* __restrict__ weight,
                          const int* __restrict__ idx, float* __restrict__ out,
                          float* __restrict__ partial) {
  __shared__ float red[256];
  int b  = blockIdx.y;
  int hw = blockIdx.x * 256 + threadIdx.x;
  int code = idx[b * HWQ + hw];
  const float* wrow = weight + (size_t)code * DQ;
  const float* lp = latent + (size_t)b * DQ * HWQ + hw;
  float* op = out + (size_t)b * DQ * HWQ + hw;
  float acc = 0.f;
#pragma unroll 4
  for (int d = 0; d < DQ; ++d) {
    float q = wrow[d];                // L2-resident codebook
    float x = lp[(size_t)d * HWQ];    // coalesced over hw
    op[(size_t)d * HWQ] = q;          // coalesced over hw
    float df = q - x;
    acc += df * df;
  }
  red[threadIdx.x] = acc;
  __syncthreads();
  for (int s = 128; s > 0; s >>= 1) {
    if (threadIdx.x < s) red[threadIdx.x] += red[threadIdx.x + s];
    __syncthreads();
  }
  if (threadIdx.x == 0) partial[blockIdx.y * gridDim.x + blockIdx.x] = red[0];
}

// ---------- stage 3: deterministic final loss reduction ----------
__global__ void vq_loss(const float* __restrict__ partial, float* __restrict__ loss) {
  float s = 0.f;
  for (int i = 0; i < 128; ++i) s += partial[i];
  *loss = s * (1.0f / (float)(BQ * DQ * HWQ));
}

extern "C" void kernel_launch(void* const* d_in, const int* in_sizes, int n_in,
                              void* d_out, int out_size, void* d_ws, size_t ws_size,
                              hipStream_t stream) {
  const float* latent = (const float*)d_in[0];  // (32,256,32,32) f32
  const float* weight = (const float*)d_in[1];  // (1024,256) f32
  float* out  = (float*)d_out;                  // (32,256,32,32) f32
  float* loss = out + (size_t)BQ * DQ * HWQ;    // scalar after out

  char* ws = (char*)d_ws;
  unsigned short* whi = (unsigned short*)(ws);              // 512 KB
  unsigned short* wlo = (unsigned short*)(ws + 524288);     // 512 KB
  float*          ek2 = (float*)(ws + 1048576);             //   4 KB
  int*           idxb = (int*)(ws + 1052672);               // 128 KB
  unsigned short* xhi = (unsigned short*)(ws + 1183744);    //  16 MB
  unsigned short* xlo = (unsigned short*)(ws + 17960960);   //  16 MB
  float*      partial = (float*)(ws + 34738176);            // 512 B

  vq_prep_weight<<<KQ, 256, 0, stream>>>(weight, whi, wlo, ek2);
  vq_prep_latent<<<dim3(HWQ / 32, DQ / 32, BQ), dim3(32, 8), 0, stream>>>(latent, xhi, xlo);
  vq_argmin<<<NQ / 128, 256, SMEM_BYTES, stream>>>(xhi, xlo, whi, wlo, ek2, idxb);
  vq_gather<<<dim3(HWQ / 256, BQ), 256, 0, stream>>>(latent, weight, idxb, out, partial);
  vq_loss<<<1, 1, 0, stream>>>(partial, loss);
}
